// SelfAttention_27943057228093
// MI455X (gfx1250) — matrix operand order repaired and checked by reading, not compiled, not run on hardware
//
#include <hip/hip_runtime.h>
#include <hip/hip_bf16.h>

// ---------------------------------------------------------------------------
// CDNA5 (gfx1250) wave32 WMMA cross-attention.
//   x  : (8,128,32,32) f32     xe : (8,128,64,64) f32
//   q_w: (128,128)  kv_w: (256,128)  out_w: (128,128)  out_b: (128,)
// Pipeline: qkv projections (bf16 out) -> fused flash attention -> out proj.
// All matmuls via v_wmma_f32_16x16x32_bf16 (bf16 A/B, f32 accum, K=32 = head dim).
// ---------------------------------------------------------------------------

typedef __attribute__((ext_vector_type(16))) __bf16 v16bf;
typedef __attribute__((ext_vector_type(8)))  float  v8f;

union BFrag {
    v16bf v;
    unsigned int u[8];
};

__device__ __forceinline__ unsigned short f2bf(float f) {
    union { __hip_bfloat16 h; unsigned short u; } c;
    c.h = __float2bfloat16(f);
    return c.u;
}

__device__ __forceinline__ unsigned int pkbf2(float lo, float hi) {
    union { __hip_bfloat162 h; unsigned int u; } c;
    c.h = __float22bfloat162_rn(float2{lo, hi});
    return c.u;
}

#define WMMA_BF16(A, B, C) \
    __builtin_amdgcn_wmma_f32_16x16x32_bf16(false, (A), false, (B), (short)0, (C), false, false)

// ---------------------------------------------------------------------------
// Kernel 1: projection  W(Mx128) * X[b](128 x P) -> bf16, scatter per mode.
// MODE 0: Q  -> dstA[bn][p][d]   (pre-scaled by 1/sqrt(C))
// MODE 1: KV -> blockIdx.y==0: K[bn][s][d];  blockIdx.y==1: V[bn][d][s]
// The K/V split is uniform per block (decided by blockIdx.y), so the store
// scatter has no divergence.
// ---------------------------------------------------------------------------
template <int MODE>
__global__ __launch_bounds__(256)
void proj_qkv_kernel(const float* __restrict__ W,
                     const float* __restrict__ X,
                     int P,
                     unsigned short* __restrict__ dstA,
                     unsigned short* __restrict__ dstV,
                     float oscale)
{
    __shared__ unsigned short ldsX[64 * 130];  // [p_local][c] bf16, padded stride

    const int tid   = threadIdx.x;
    const int lane  = tid & 31;
    const int wave  = tid >> 5;
    const int l15   = lane & 15;
    const int lhalf = lane >> 4;
    const int b     = blockIdx.z;
    const int p0    = blockIdx.x * 64;

    const float* Xb = X + (size_t)b * 128 * P;

    // Stage 128(C) x 64(P) f32 tile -> LDS transposed bf16 (coalesced reads)
    #pragma unroll
    for (int it = 0; it < 32; ++it) {
        int i  = tid + 256 * it;
        int c  = i >> 6;
        int pl = i & 63;
        ldsX[pl * 130 + c] = f2bf(Xb[(size_t)c * P + p0 + pl]);
    }
    __syncthreads();

    // A fragments (weights): 4 K-chunks of 32, held in registers
    const int rowA = blockIdx.y * 128 + wave * 16 + l15;
    const float* Wr = W + (size_t)rowA * 128;
    const int klo = lhalf ? 16 : 0;   // A layout: lanes<16 -> K 0-15, else 16-31
    BFrag a[4];
    #pragma unroll
    for (int kk = 0; kk < 4; ++kk) {
        int kc = kk * 32;
        #pragma unroll
        for (int j = 0; j < 4; ++j) {
            int k0 = kc + klo + 2 * j;
            int k1 = kc + klo + 8 + 2 * j;
            a[kk].u[j]     = pkbf2(Wr[k0], Wr[k0 + 1]);
            a[kk].u[j + 4] = pkbf2(Wr[k1], Wr[k1 + 1]);
        }
    }

    const bool isV = (MODE == 1) && (blockIdx.y != 0);   // uniform per block
    const int kh = lhalf ? 16 : 0;   // B layout: lanes<16 -> K 0-15, else 16-31
    #pragma unroll
    for (int sub = 0; sub < 4; ++sub) {
        const int pl = sub * 16 + l15;
        const unsigned int* row32 = (const unsigned int*)(ldsX + pl * 130);
        v8f acc = {};
        #pragma unroll
        for (int kk = 0; kk < 4; ++kk) {
            BFrag bfr;
            const int base = (kk * 32 + kh) >> 1;
            #pragma unroll
            for (int j = 0; j < 8; ++j) bfr.u[j] = row32[base + j];
            acc = WMMA_BF16(a[kk].v, bfr.v, acc);
        }
        // Scatter D (C-layout: vgpr g -> row g or g+8 by lane half, col = l15)
        const int pcol = p0 + sub * 16 + l15;
        #pragma unroll
        for (int g = 0; g < 8; ++g) {
            const int m  = wave * 16 + g + 8 * lhalf;      // 0..127 within block
            const int n  = m >> 5, d = m & 31;
            const int bn = b * 4 + n;
            const unsigned short bv = f2bf(acc[g] * oscale);
            if (MODE == 0) {
                dstA[((size_t)bn * 1024 + pcol) * 32 + d] = bv;
            } else if (!isV) {
                dstA[((size_t)bn * 4096 + pcol) * 32 + d] = bv;
            } else {
                dstV[((size_t)bn * 32 + d) * 4096 + pcol] = bv;
            }
        }
    }
}

// ---------------------------------------------------------------------------
// Kernel 2: fused flash attention per (b, head). Block = 8 waves = 128
// queries. K/V streamed in 64-key chunks staged cooperatively into LDS
// (4 KB + 4 KB per chunk), then per wave: 4 WMMA QK^T + online softmax +
// 4 WMMA AV (two 32-key halves through the per-wave LDS P-staging slice).
// ---------------------------------------------------------------------------
__global__ __launch_bounds__(256)
void attn_kernel(const unsigned short* __restrict__ q,
                 const unsigned short* __restrict__ k,
                 const unsigned short* __restrict__ v,
                 unsigned short* __restrict__ ao)
{
    __shared__ unsigned short sK[64 * 32];     // [s_local][d]
    __shared__ unsigned short sV[32 * 64];     // [d][s_local]
    __shared__ unsigned short sP[8][16 * 32];  // per-wave exp(S) staging

    const int tid   = threadIdx.x;
    const int lane  = tid & 31;
    const int wave  = tid >> 5;
    const int l15   = lane & 15;
    const int lhalf = lane >> 4;
    const int b     = blockIdx.z;
    const int n     = blockIdx.y;
    const int bn    = b * 4 + n;
    const int q0    = blockIdx.x * 128 + wave * 16;

    // Q A-fragment: q stored [bn][p][d] bf16, d pairs -> dword loads
    const unsigned int* q32 = (const unsigned int*)q;
    const int klo2 = lhalf ? 8 : 0;   // dword offset of A-lane K chunk
    BFrag aQ;
    {
        const size_t base = ((size_t)bn * 1024 + (q0 + l15)) * 16;
        #pragma unroll
        for (int j = 0; j < 4; ++j) {
            aQ.u[j]     = q32[base + klo2 + j];
            aQ.u[j + 4] = q32[base + klo2 + 4 + j];
        }
    }

    float m_i[8], l_i[8];
    v8f acc0 = {}, acc1 = {};
    #pragma unroll
    for (int g = 0; g < 8; ++g) { m_i[g] = -3.0e38f; l_i[g] = 0.0f; }

    const unsigned short* kbase = k + (size_t)bn * 4096 * 32;
    const unsigned short* vbase = v + (size_t)bn * 32 * 4096;
    const int khB = lhalf ? 16 : 0;
    const int vrow = tid >> 3;           // V staging: row d
    const int vseg = tid & 7;            // 16B segment within 64-key row

    for (int s0 = 0; s0 < 4096; s0 += 64) {
        // ---- cooperative staging: K chunk (contiguous 4 KB), V chunk (32x128B)
        {
            const uint4* ksrc = (const uint4*)(kbase + (size_t)s0 * 32);
            ((uint4*)sK)[tid] = ksrc[tid];
            const uint4* vsrc = (const uint4*)(vbase + (size_t)vrow * 4096 + s0 + vseg * 8);
            ((uint4*)sV)[tid] = vsrc[0];
            if (s0 + 64 < 4096) {
                __builtin_prefetch(kbase + (size_t)(s0 + 64) * 32 + tid * 8, 0, 1);
                __builtin_prefetch(vbase + (size_t)vrow * 4096 + s0 + 64 + vseg * 8, 0, 1);
            }
        }
        __syncthreads();

        // ---- QK^T: 4 x 16-key tiles (K dim = full head dim 32)
        const unsigned int* sK32 = (const unsigned int*)sK;
        v8f st[4];
        #pragma unroll
        for (int t = 0; t < 4; ++t) {
            BFrag bk;
            const int base = ((t * 16 + l15) * 32 + khB) >> 1;
            #pragma unroll
            for (int j = 0; j < 8; ++j) bk.u[j] = sK32[base + j];
            v8f z = {};
            st[t] = WMMA_BF16(aQ.v, bk.v, z);
        }

        // ---- online softmax (rows live across 16-lane halves; xor 1|2|4|8)
        float pe[4][8], alpha[8];
        #pragma unroll
        for (int g = 0; g < 8; ++g) {
            float cand = fmaxf(fmaxf(st[0][g], st[1][g]), fmaxf(st[2][g], st[3][g]));
            cand = fmaxf(cand, __shfl_xor(cand, 1, 32));
            cand = fmaxf(cand, __shfl_xor(cand, 2, 32));
            cand = fmaxf(cand, __shfl_xor(cand, 4, 32));
            cand = fmaxf(cand, __shfl_xor(cand, 8, 32));
            const float mn = fmaxf(m_i[g], cand);
            alpha[g] = __expf(m_i[g] - mn);
            m_i[g]   = mn;
            float rs = 0.0f;
            #pragma unroll
            for (int t = 0; t < 4; ++t) {
                const float e = __expf(st[t][g] - mn);
                pe[t][g] = e;
                rs += e;
            }
            rs += __shfl_xor(rs, 1, 32);
            rs += __shfl_xor(rs, 2, 32);
            rs += __shfl_xor(rs, 4, 32);
            rs += __shfl_xor(rs, 8, 32);
            l_i[g] = l_i[g] * alpha[g] + rs;
        }
        #pragma unroll
        for (int g = 0; g < 8; ++g) { acc0[g] *= alpha[g]; acc1[g] *= alpha[g]; }

        // ---- AV in two 32-key halves via per-wave LDS re-layout
        const unsigned int* sV32 = (const unsigned int*)sV;
        unsigned short* sp = sP[wave];
        #pragma unroll
        for (int h = 0; h < 2; ++h) {
            #pragma unroll
            for (int g = 0; g < 8; ++g) {
                const int r = g + 8 * lhalf;
                sp[r * 32 + l15]      = f2bf(pe[2 * h][g]);
                sp[r * 32 + 16 + l15] = f2bf(pe[2 * h + 1][g]);
            }
            __builtin_amdgcn_wave_barrier();   // DS ops in-order within a wave
            BFrag aP;
            {
                const unsigned int* row32 = (const unsigned int*)(sp + l15 * 32);
                #pragma unroll
                for (int j = 0; j < 4; ++j) {
                    aP.u[j]     = row32[klo2 + j];
                    aP.u[j + 4] = row32[klo2 + 4 + j];
                }
            }
            __builtin_amdgcn_wave_barrier();
            const int kb = h * 32 + khB;       // key offset within the 64-chunk
            BFrag bv;
            int base = (l15 * 64 + kb) >> 1;
            #pragma unroll
            for (int j = 0; j < 8; ++j) bv.u[j] = sV32[base + j];
            acc0 = WMMA_BF16(aP.v, bv.v, acc0);
            base = ((16 + l15) * 64 + kb) >> 1;
            #pragma unroll
            for (int j = 0; j < 8; ++j) bv.u[j] = sV32[base + j];
            acc1 = WMMA_BF16(aP.v, bv.v, acc1);
        }
        __syncthreads();   // protect sK/sV before next chunk's staging
    }

    // ---- epilogue: normalize and store bf16 as [b][p][c] (c = n*32 + d)
    #pragma unroll
    for (int g = 0; g < 8; ++g) {
        const float inv = 1.0f / l_i[g];
        const int p = q0 + g + 8 * lhalf;
        const size_t base = ((size_t)b * 1024 + p) * 128 + n * 32;
        ao[base + l15]      = f2bf(acc0[g] * inv);
        ao[base + 16 + l15] = f2bf(acc1[g] * inv);
    }
}

// ---------------------------------------------------------------------------
// Kernel 3: out_w(128x128) * AO[b](128 x 1024) + bias + residual -> f32 out.
// AO already bf16 [b][p][c] so B-fragments are contiguous dword loads.
// ---------------------------------------------------------------------------
__global__ __launch_bounds__(256)
void out_proj_kernel(const float* __restrict__ Wo,
                     const unsigned short* __restrict__ ao,
                     const float* __restrict__ bias,
                     const float* __restrict__ x,
                     float* __restrict__ out)
{
    const int tid   = threadIdx.x;
    const int lane  = tid & 31;
    const int wave  = tid >> 5;
    const int l15   = lane & 15;
    const int lhalf = lane >> 4;
    const int b     = blockIdx.z;
    const int p0    = blockIdx.x * 64;

    const int rowA = wave * 16 + l15;
    const float* Wr = Wo + (size_t)rowA * 128;
    const int klo = lhalf ? 16 : 0;
    BFrag a[4];
    #pragma unroll
    for (int kk = 0; kk < 4; ++kk) {
        int kc = kk * 32;
        #pragma unroll
        for (int j = 0; j < 4; ++j) {
            int k0 = kc + klo + 2 * j;
            int k1 = kc + klo + 8 + 2 * j;
            a[kk].u[j]     = pkbf2(Wr[k0], Wr[k0 + 1]);
            a[kk].u[j + 4] = pkbf2(Wr[k1], Wr[k1 + 1]);
        }
    }

    const unsigned int* ao32 = (const unsigned int*)ao;
    const int kh = lhalf ? 16 : 0;
    #pragma unroll
    for (int sub = 0; sub < 4; ++sub) {
        const int p = p0 + sub * 16 + l15;
        const size_t rbase = ((size_t)b * 1024 + p) * 64;   // dwords per row = 64
        v8f acc = {};
        #pragma unroll
        for (int kk = 0; kk < 4; ++kk) {
            BFrag bfr;
            const int base = (kk * 32 + kh) >> 1;
            #pragma unroll
            for (int j = 0; j < 8; ++j) bfr.u[j] = ao32[rbase + base + j];
            acc = WMMA_BF16(a[kk].v, bfr.v, acc);
        }
        #pragma unroll
        for (int g = 0; g < 8; ++g) {
            const int o = wave * 16 + g + 8 * lhalf;
            const size_t idx = ((size_t)b * 128 + o) * 1024 + p;
            out[idx] = acc[g] + bias[o] + x[idx];
        }
    }
}

// ---------------------------------------------------------------------------
extern "C" void kernel_launch(void* const* d_in, const int* in_sizes, int n_in,
                              void* d_out, int out_size, void* d_ws, size_t ws_size,
                              hipStream_t stream)
{
    const float* x     = (const float*)d_in[0];   // (8,128,32,32)
    const float* xe    = (const float*)d_in[1];   // (8,128,64,64)
    const float* q_w   = (const float*)d_in[2];   // (128,128)
    const float* kv_w  = (const float*)d_in[3];   // (256,128)
    const float* out_w = (const float*)d_in[4];   // (128,128)
    const float* out_b = (const float*)d_in[5];   // (128,)
    float* out = (float*)d_out;

    // Workspace layout (bf16): q 2MB | k 8MB | v 8MB | ao 2MB  (total 20 MB)
    unsigned short* ws    = (unsigned short*)d_ws;
    unsigned short* q_bf  = ws;
    unsigned short* k_bf  = q_bf + (size_t)8 * 4 * 1024 * 32;
    unsigned short* v_bf  = k_bf + (size_t)8 * 4 * 4096 * 32;
    unsigned short* ao_bf = v_bf + (size_t)8 * 4 * 4096 * 32;

    const float qscale = 0.08838834764831845f;   // 1/sqrt(128)

    proj_qkv_kernel<0><<<dim3(16, 1, 8), 256, 0, stream>>>(q_w, x, 1024, q_bf, nullptr, qscale);
    proj_qkv_kernel<1><<<dim3(64, 2, 8), 256, 0, stream>>>(kv_w, xe, 4096, k_bf, v_bf, 1.0f);
    attn_kernel<<<dim3(8, 4, 8), 256, 0, stream>>>(q_bf, k_bf, v_bf, ao_bf);
    out_proj_kernel<<<dim3(16, 1, 8), 256, 0, stream>>>(out_w, ao_bf, out_b, x, out);
}
